// TransformerConv2_17188459119338
// MI455X (gfx1250) — compile-verified, matrix operand order checked by reading, not run on hardware
//
#include <hip/hip_runtime.h>
#include <hip/hip_bf16.h>

// ---------------------------------------------------------------------------
// Graph TransformerConv x3 + mean-pool + linear + softmax for MI455X (gfx1250)
// Dense projections: fused [fin x 256] bf16 WMMA GEMM (v_wmma_f32_16x16x32_bf16)
//   - A tile (bf16) staged via GLOBAL_LOAD_ASYNC_TO_LDS_B128, fragments from
//     contiguous ds_load_b128 (no conversion VALU in the hot loop)
//   - B matrix pre-swizzled to fragment order -> 2x global_load_b128 per frag
// Sparse attention: fp32 VALU + L2-resident float atomics
// ---------------------------------------------------------------------------

typedef __attribute__((ext_vector_type(16))) __bf16          v16bf;
typedef __attribute__((ext_vector_type(16))) unsigned short  v16s;
typedef __attribute__((ext_vector_type(8)))  unsigned short  v8s;
typedef __attribute__((ext_vector_type(8)))  float           v8f;

#define HEADS 4
#define CH    16
#define HC    64      // HEADS*CH
#define QK4   256     // 4 fused matrices * HC

#if defined(__gfx1250__) && __has_builtin(__builtin_amdgcn_global_load_async_to_lds_b128) && \
    __has_builtin(__builtin_amdgcn_s_wait_asynccnt)
#define USE_ASYNC_LDS 1
typedef int v4i __attribute__((vector_size(16)));
typedef __attribute__((address_space(1))) v4i GV4;   // global int4*  (param 1)
typedef __attribute__((address_space(3))) v4i LV4;   // LDS int4*     (param 2)
#else
#define USE_ASYNC_LDS 0
#endif

static __device__ __forceinline__ unsigned short f2bf(float f) {
  unsigned int u = __float_as_uint(f);
  u += 0x7FFFu + ((u >> 16) & 1u);           // round-to-nearest-even
  return (unsigned short)(u >> 16);
}
// monotonic float <-> uint mapping so segment-max uses global_atomic_max_u32
static __device__ __forceinline__ unsigned int fmono(float f) {
  unsigned int b = __float_as_uint(f);
  return (b & 0x80000000u) ? ~b : (b | 0x80000000u);
}
static __device__ __forceinline__ float funmono(unsigned int u) {
  unsigned int b = (u & 0x80000000u) ? (u & 0x7FFFFFFFu) : ~u;
  return __uint_as_float(b);
}

// ---------------- f32 -> bf16 streaming convert (for x) ---------------------
__global__ void cvt_bf16_kernel(const float* __restrict__ in, unsigned short* __restrict__ out,
                                long long n) {
  long long i = (long long)blockIdx.x * blockDim.x + threadIdx.x;
  if (i < n) out[i] = f2bf(in[i]);
}

// ---------------- weight prep ------------------------------------------------
// Writes Wcat in FRAGMENT order: for col-tile ct (16), k-step ks (fin/32),
// lane (32): 16 contiguous bf16 = that lane's B operand for one WMMA.
// B-fragment element j (=2v+odd) of lane {half,nIdx}:
//   k = ks*32 + (v>>2)*16 + half*8 + (v&3)*2 + odd ; n = ct*16 + nIdx
__global__ void prep_w_kernel(const float* __restrict__ Wq, const float* __restrict__ Wk,
                              const float* __restrict__ Wv, const float* __restrict__ Ws,
                              const float* __restrict__ bq, const float* __restrict__ bk,
                              const float* __restrict__ bv, const float* __restrict__ bs,
                              unsigned short* __restrict__ Wcat, float* __restrict__ bcat,
                              int fin) {
  int i = blockIdx.x * blockDim.x + threadIdx.x;
  int total = fin * QK4;
  int nks = fin >> 5;
  if (i < total) {
    int j    = i & 15;
    int lane = (i >> 4) & 31;
    int t2   = i >> 9;              // ct*nks + ks
    int ks   = t2 % nks;
    int ct   = t2 / nks;
    int half = lane >> 4, nIdx = lane & 15;
    int v = j >> 1, odd = j & 1;
    int k = (ks << 5) + ((v >> 2) << 4) + (half << 3) + ((v & 3) << 1) + odd;
    int c = ct * 16 + nIdx;
    int m = c >> 6, cc = c & (HC - 1);
    const float* W = (m == 0) ? Wq : (m == 1) ? Wk : (m == 2) ? Wv : Ws;
    Wcat[i] = f2bf(W[k * HC + cc]);
  }
  if (i < QK4) {
    int m = i >> 6, cc = i & (HC - 1);
    const float* b = (m == 0) ? bq : (m == 1) ? bk : (m == 2) ? bv : bs;
    bcat[i] = b[cc];
  }
}

// ---------------- fused QKVS projection GEMM (WMMA bf16, f32 accum) ---------
// Out[N x 256] = Xbf[N x FIN] @ W + b ; 256 thr = 8 waves; block owns 16 rows;
// each wave computes two 16x16 output tiles, A fragments cached in registers.
template <int FIN>
__global__ __launch_bounds__(256)
void gemm_qkvs_kernel(const unsigned short* __restrict__ Xbf,
                      const unsigned short* __restrict__ Wfrag,
                      const float* __restrict__ bcat, float* __restrict__ Out, int Ntot) {
  constexpr int NKS = FIN / 32;
  __shared__ unsigned short Xs[16 * FIN];        // bf16 row tile (4KB @ FIN=128)
  const int tid = threadIdx.x;
  const int m0  = blockIdx.x * 16;
  const bool full = (m0 + 16 <= Ntot);

  if (full) {
#if USE_ASYNC_LDS
#pragma unroll
    for (int i = tid; i < 16 * FIN / 8; i += 256) {
      const unsigned short* g = Xbf + (size_t)m0 * FIN + i * 8;
      __builtin_amdgcn_global_load_async_to_lds_b128((GV4*)g, (LV4*)&Xs[i * 8], 0, 0);
    }
    __builtin_amdgcn_s_wait_asynccnt(0);
#else
#pragma unroll
    for (int i = tid; i < 16 * FIN / 8; i += 256)
      *(v8s*)&Xs[i * 8] = *(const v8s*)(Xbf + (size_t)m0 * FIN + i * 8);
#endif
  } else {
    for (int i = tid; i < 16 * FIN; i += 256) {
      int r = i / FIN, c = i - r * FIN;
      Xs[i] = (m0 + r < Ntot) ? Xbf[(size_t)(m0 + r) * FIN + c] : (unsigned short)0;
    }
  }
  __syncthreads();

  const int wave = tid >> 5, lane = tid & 31;
  const int half = lane >> 4, mn = lane & 15;    // A: M index / B,D: N index

  // Build all A fragments once: two contiguous 16B LDS reads each, no VALU cvt
  v16s afr[NKS];
#pragma unroll
  for (int ks = 0; ks < NKS; ++ks) {
    const unsigned short* ar = &Xs[mn * FIN + ks * 32 + half * 8];
    v8s lo = *(const v8s*)ar;          // K = ks*32 + half*8 + 0..7
    v8s hi = *(const v8s*)(ar + 16);   // K = ks*32 + 16 + half*8 + 0..7
    afr[ks] = __builtin_shufflevector(lo, hi, 0, 1, 2, 3, 4, 5, 6, 7,
                                      8, 9, 10, 11, 12, 13, 14, 15);
  }

#pragma unroll
  for (int t = 0; t < 2; ++t) {
    const int ct = wave * 2 + t;                 // column tile 0..15
    v8f acc = {};
#pragma unroll
    for (int ks = 0; ks < NKS; ++ks) {
      const v16s* bp = (const v16s*)(Wfrag + (((size_t)ct * NKS + ks) * 32 + lane) * 16);
      acc = __builtin_amdgcn_wmma_f32_16x16x32_bf16(
                false, __builtin_bit_cast(v16bf, afr[ks]),
                false, __builtin_bit_cast(v16bf, *bp),
                (short)0, acc, false, false);
    }
    const int col = ct * 16 + mn;
    const float bb = bcat[col];
    if (full) {
#pragma unroll
      for (int r = 0; r < 8; ++r)
        Out[(size_t)(m0 + r + half * 8) * QK4 + col] = acc[r] + bb;
    } else {
      for (int r = 0; r < 8; ++r) {
        int M = r + half * 8;
        if (m0 + M < Ntot) Out[(size_t)(m0 + M) * QK4 + col] = acc[r] + bb;
      }
    }
  }
}

// ---------------- zero fill (u32) -------------------------------------------
__global__ void fill_zero_kernel(unsigned int* __restrict__ p, long long n) {
  long long i = (long long)blockIdx.x * blockDim.x + threadIdx.x;
  if (i < n) p[i] = 0u;
}

// ---------------- edge pass 1: scores + segment max -------------------------
__global__ void edge_scores_kernel(const float* __restrict__ QKVS,
                                   const int* __restrict__ src, const int* __restrict__ dst,
                                   float* __restrict__ score, unsigned int* __restrict__ maxb,
                                   int Etot) {
  int idx = blockIdx.x * blockDim.x + threadIdx.x;
  if (idx >= Etot * HEADS) return;
  int e = idx >> 2, h = idx & 3;
  int s = src[e], d = dst[e];
  const float4* qp = (const float4*)(QKVS + (size_t)d * QK4 + h * CH);        // Q of dst
  const float4* kp = (const float4*)(QKVS + (size_t)s * QK4 + HC + h * CH);   // K of src
  // warm L2 for the scatter pass (V row of src) -> global_prefetch_b8
  __builtin_prefetch(QKVS + (size_t)s * QK4 + 2 * HC + h * CH, 0, 1);
  float acc = 0.0f;
#pragma unroll
  for (int i = 0; i < 4; ++i) {
    float4 q = qp[i], k = kp[i];
    acc += q.x * k.x + q.y * k.y + q.z * k.z + q.w * k.w;
  }
  acc *= 0.25f;                                    // 1/sqrt(C), C = 16
  score[idx] = acc;
  atomicMax(maxb + (size_t)d * HEADS + h, fmono(acc));
}

// ---------------- edge pass 2: exp, denom, unnormalized p*v scatter ---------
__global__ void edge_accum_kernel(const float* __restrict__ QKVS,
                                  const int* __restrict__ src, const int* __restrict__ dst,
                                  const float* __restrict__ score,
                                  const unsigned int* __restrict__ maxb,
                                  float* __restrict__ denom, float* __restrict__ agg,
                                  int Etot) {
  int idx = blockIdx.x * blockDim.x + threadIdx.x;
  if (idx >= Etot * HEADS) return;
  int e = idx >> 2, h = idx & 3;
  int s = src[e], d = dst[e];
  float m = funmono(maxb[(size_t)d * HEADS + h]);
  float p = __expf(score[idx] - m);
  atomicAdd(denom + (size_t)d * HEADS + h, p);
  const float4* vp = (const float4*)(QKVS + (size_t)s * QK4 + 2 * HC + h * CH); // V of src
  float* ap = agg + (size_t)d * HC + h * CH;
#pragma unroll
  for (int i = 0; i < 4; ++i) {
    float4 vv = vp[i];
    atomicAdd(ap + i * 4 + 0, p * vv.x);
    atomicAdd(ap + i * 4 + 1, p * vv.y);
    atomicAdd(ap + i * 4 + 2, p * vv.z);
    atomicAdd(ap + i * 4 + 3, p * vv.w);
  }
}

// ---------------- node finalize: normalize + skip (+ReLU) -------------------
// Writes f32 (for pooling / inspection) and bf16 (feed for next layer's GEMM)
__global__ void node_finalize_kernel(const float* __restrict__ QKVS,
                                     const float* __restrict__ agg,
                                     const float* __restrict__ denom,
                                     float* __restrict__ Hout,
                                     unsigned short* __restrict__ Hbf,
                                     int relu, int Ntot) {
  int i = blockIdx.x * blockDim.x + threadIdx.x;
  if (i >= Ntot * HC) return;
  int n = i >> 6, j = i & (HC - 1), h = j >> 4;
  float v = agg[i] / (denom[(size_t)n * HEADS + h] + 1e-16f)
          + QKVS[(size_t)n * QK4 + 3 * HC + j];                // S (skip) part
  if (relu) v = fmaxf(v, 0.0f);
  Hout[i] = v;
  Hbf[i]  = f2bf(v);
}

// ---------------- global mean pool (accumulate) -----------------------------
__global__ void pool_accum_kernel(const float* __restrict__ Hf, const int* __restrict__ batch,
                                  float* __restrict__ pooled, float* __restrict__ counts,
                                  int Ntot) {
  int i = blockIdx.x * blockDim.x + threadIdx.x;
  if (i >= Ntot * HC) return;
  int n = i >> 6, j = i & (HC - 1);
  int g = batch[n];
  atomicAdd(pooled + (size_t)g * HC + j, Hf[i]);
  if (j == 0) atomicAdd(counts + g, 1.0f);
}

// ---------------- classifier + softmax --------------------------------------
__global__ void classify_kernel(const float* __restrict__ pooled, const float* __restrict__ counts,
                                const float* __restrict__ Wl, const float* __restrict__ bl,
                                float* __restrict__ out, int G, int ncls) {
  int g = blockIdx.x * blockDim.x + threadIdx.x;
  if (g >= G) return;
  float inv = 1.0f / fmaxf(counts[g], 1.0f);
  float logits[16];
  float mx = -3.4e38f;
  for (int c = 0; c < ncls; ++c) {
    float acc = bl[c];
    for (int j = 0; j < HC; ++j) acc += pooled[(size_t)g * HC + j] * inv * Wl[j * ncls + c];
    logits[c] = acc;
    mx = fmaxf(mx, acc);
  }
  float sum = 0.0f;
  for (int c = 0; c < ncls; ++c) { logits[c] = __expf(logits[c] - mx); sum += logits[c]; }
  float isum = 1.0f / sum;
  for (int c = 0; c < ncls; ++c) out[(size_t)g * ncls + c] = logits[c] * isum;
}

// ---------------------------------------------------------------------------
extern "C" void kernel_launch(void* const* d_in, const int* in_sizes, int n_in,
                              void* d_out, int out_size, void* d_ws, size_t ws_size,
                              hipStream_t stream) {
  const float* x     = (const float*)d_in[0];
  const int*   eidx  = (const int*)d_in[1];
  const int*   batch = (const int*)d_in[2];

  const int N    = in_sizes[2];
  const int E    = in_sizes[1] / 2;
  const int F    = in_sizes[0] / N;            // 128
  const int NCLS = in_sizes[28];               // 10
  const int G    = out_size / NCLS;            // 128
  if (F != 128) return;                        // reference shape is fixed

  const int* src = eidx;
  const int* dst = eidx + E;

  const float* W_lin = (const float*)d_in[27];
  const float* b_lin = (const float*)d_in[28];

  // workspace carve-up (256B aligned sections)
  char* base = (char*)d_ws;
  size_t off = 0;
  auto carve = [&](size_t bytes) -> char* {
    char* p = base + off;
    off = (off + bytes + 255) & ~(size_t)255;
    return p;
  };
  const int fins[3] = {F, HC, HC};
  unsigned short* Wcat[3];
  float* bcat[3];
  for (int l = 0; l < 3; ++l) {
    Wcat[l] = (unsigned short*)carve((size_t)fins[l] * QK4 * sizeof(unsigned short));
    bcat[l] = (float*)carve(QK4 * sizeof(float));
  }
  unsigned short* Xbf   = (unsigned short*)carve((size_t)N * F * sizeof(unsigned short));
  unsigned short* Hbf   = (unsigned short*)carve((size_t)N * HC * sizeof(unsigned short));
  float*        QKVS   = (float*)carve((size_t)N * QK4 * sizeof(float));
  float*        score  = (float*)carve((size_t)E * HEADS * sizeof(float));
  float*        agg    = (float*)carve((size_t)N * HC * sizeof(float));      // agg+denom+maxb
  float*        denom  = (float*)carve((size_t)N * HEADS * sizeof(float));   //  zeroed together
  unsigned int* maxb   = (unsigned int*)carve((size_t)N * HEADS * sizeof(unsigned int));
  float*        Hbuf   = (float*)carve((size_t)N * HC * sizeof(float));
  float*        pooled = (float*)carve((size_t)G * HC * sizeof(float));
  float*        counts = (float*)carve((size_t)G * sizeof(float));
  if (off > ws_size) return;                   // workspace too small: bail safely

  const int TPB = 256;

  // convert + swizzle weights, convert x to bf16 (streaming, once per call)
  for (int l = 0; l < 3; ++l) {
    const float* const* W = (const float* const*)(d_in + 3 + l * 8);
    int wtot = fins[l] * QK4;
    prep_w_kernel<<<(wtot + TPB - 1) / TPB, TPB, 0, stream>>>(
        (const float*)W[0], (const float*)W[2], (const float*)W[4], (const float*)W[6],
        (const float*)W[1], (const float*)W[3], (const float*)W[5], (const float*)W[7],
        Wcat[l], bcat[l], fins[l]);
  }
  long long xN = (long long)N * F;
  cvt_bf16_kernel<<<(int)((xN + TPB - 1) / TPB), TPB, 0, stream>>>(x, Xbf, xN);

  const int rowTiles = (N + 15) / 16;
  const long long zeroN = (long long)N * (HC + 2 * HEADS);   // agg + denom + maxb
  const int eThreads = E * HEADS;
  const int nThreads = N * HC;

  const unsigned short* layerIn = Xbf;
  for (int l = 0; l < 3; ++l) {
    if (l == 0)
      gemm_qkvs_kernel<128><<<rowTiles, TPB, 0, stream>>>(layerIn, Wcat[l], bcat[l], QKVS, N);
    else
      gemm_qkvs_kernel<64><<<rowTiles, TPB, 0, stream>>>(layerIn, Wcat[l], bcat[l], QKVS, N);
    fill_zero_kernel<<<(int)((zeroN + TPB - 1) / TPB), TPB, 0, stream>>>((unsigned int*)agg, zeroN);
    edge_scores_kernel<<<(eThreads + TPB - 1) / TPB, TPB, 0, stream>>>(QKVS, src, dst, score, maxb, E);
    edge_accum_kernel<<<(eThreads + TPB - 1) / TPB, TPB, 0, stream>>>(QKVS, src, dst, score, maxb,
                                                                      denom, agg, E);
    node_finalize_kernel<<<(nThreads + TPB - 1) / TPB, TPB, 0, stream>>>(QKVS, agg, denom, Hbuf,
                                                                         Hbf, (l < 2) ? 1 : 0, N);
    layerIn = Hbf;
  }

  long long poolN = (long long)G * HC + G;
  fill_zero_kernel<<<(int)((poolN + TPB - 1) / TPB), TPB, 0, stream>>>((unsigned int*)pooled, poolN);
  pool_accum_kernel<<<(nThreads + TPB - 1) / TPB, TPB, 0, stream>>>(Hbuf, batch, pooled, counts, N);
  classify_kernel<<<(G + 127) / 128, 128, 0, stream>>>(pooled, counts, W_lin, b_lin,
                                                       (float*)d_out, G, NCLS);
}
